// Deepseekv3MoE_75763223102190
// MI455X (gfx1250) — matrix operand order repaired and checked by reading, not compiled, not run on hardware
//
#include <hip/hip_runtime.h>

#ifndef USE_ASYNC_LDS
#define USE_ASYNC_LDS 1
#endif
#ifndef USE_TDM
#define USE_TDM 1
#endif

#if USE_TDM && __has_builtin(__builtin_amdgcn_tensor_load_to_lds) && __has_builtin(__builtin_amdgcn_s_wait_tensorcnt)
#define HAVE_TDM 1
#else
#define HAVE_TDM 0
#endif

// ---------------- problem constants ----------------
constexpr int E  = 32;
constexpr int H  = 1024;
constexpr int I  = 512;
constexpr int IS = 2048;
constexpr int G  = 8;
constexpr int TOPK_GROUP = 4;
constexpr int TOP_K = 8;
constexpr int T  = 1024;
#define ROUTED_SCALE 2.5f

// ---------------- workspace layout (bytes) ----------------
constexpr size_t OFF_CNT  = 0;                                        // E ints
constexpr size_t OFF_TIDX = 256;                                      // E*T ints
constexpr size_t OFF_TW   = OFF_TIDX + (size_t)E * T * 4;             // E*T floats
constexpr size_t OFF_ACTR = (OFF_TW + (size_t)E * T * 4 + 255) & ~(size_t)255; // E*T*I bf16
constexpr size_t OFF_ACTS = OFF_ACTR + (size_t)E * T * I * 2;         // T*IS bf16

// ---------------- types / helpers ----------------
typedef __bf16 v16bf  __attribute__((ext_vector_type(16)));
typedef __bf16 bf16x2 __attribute__((ext_vector_type(2)));
typedef __bf16 bf16x4 __attribute__((ext_vector_type(4)));
typedef float  v8f    __attribute__((ext_vector_type(8)));
typedef float  f32x2  __attribute__((ext_vector_type(2)));
typedef float  f32x4  __attribute__((ext_vector_type(4)));
typedef unsigned int u32x4 __attribute__((ext_vector_type(4)));
typedef int          i32x4 __attribute__((ext_vector_type(4)));
typedef int          i32x8 __attribute__((ext_vector_type(8)));

union Frag {
    unsigned int u[8];
    uint4        q[2];
    v16bf        h;
};
union AccView {
    v8f   v;
    float f[8];
};

// native packed f32 -> bf16 conversions (lower to v_cvt_pk_bf16_f32 class ops)
__device__ inline unsigned pack2(float lo, float hi) {
    f32x2 f = {lo, hi};
    bf16x2 b = __builtin_convertvector(f, bf16x2);
    return __builtin_bit_cast(unsigned, b);
}
__device__ inline uint2 pack4(float4 f) {
    f32x4 v = {f.x, f.y, f.z, f.w};
    bf16x4 b = __builtin_convertvector(v, bf16x4);
    return __builtin_bit_cast(uint2, b);
}
__device__ inline unsigned short f2bf(float f) {
    __bf16 b = (__bf16)f;
    return __builtin_bit_cast(unsigned short, b);
}
__device__ inline float siluf(float x) {
    return x / (1.0f + __expf(-x));
}

// B fragment: 32(K) x 16(N) bf16, from LDS tile stored [n][k] (transposed, 32 k per row)
__device__ inline v16bf load_frag_b(const unsigned short* lB, int n_l, int kb) {
    Frag f;
    f.q[0] = *(const uint4*)(lB + n_l * 32 + kb / 2);
    f.q[1] = *(const uint4*)(lB + n_l * 32 + 16 + kb / 2);
    return f.h;
}
// A fragment from packed-bf16 uint LDS tile [r][16 uints]
__device__ inline v16bf load_frag_a_u32(const unsigned int* lA, int r, int lanehi) {
    Frag f;
    f.q[0] = *(const uint4*)(lA + r * 16 + lanehi * 4);
    f.q[1] = *(const uint4*)(lA + r * 16 + 8 + lanehi * 4);
    return f.h;
}
// A fragment from bf16(ushort) LDS tile [r][32]
__device__ inline v16bf load_frag_a_u16(const unsigned short* lA, int r, int lanehi) {
    Frag f;
    f.q[0] = *(const uint4*)(lA + r * 32 + lanehi * 8);
    f.q[1] = *(const uint4*)(lA + r * 32 + 16 + lanehi * 8);
    return f.h;
}

// ---------------- kernel 0: init ----------------
__global__ void moe_init_kernel(int* cnt) {
    if (threadIdx.x < E) cnt[threadIdx.x] = 0;
}

// ---------------- kernel 1: gate + routing ----------------
// one wave32 per token, lane == expert (E == 32 == wave32)
__global__ void gate_route_kernel(const float* __restrict__ x,
                                  const float* __restrict__ gate_w,
                                  const float* __restrict__ e_bias,
                                  int* __restrict__ cnt,
                                  int* __restrict__ tok_idx,
                                  float* __restrict__ tok_w) {
    const int lane = threadIdx.x & 31;
    const int wv   = threadIdx.x >> 5;
    const int t    = blockIdx.x * 8 + wv;

    const float* xr = x + (size_t)t * H;
    const float* wr = gate_w + (size_t)lane * H;
    float acc = 0.f;
#pragma unroll 4
    for (int h = 0; h < H; h += 4) {
        float4 xv = *(const float4*)(xr + h);
        float4 wv4 = *(const float4*)(wr + h);
        acc += xv.x * wv4.x + xv.y * wv4.y + xv.z * wv4.z + xv.w * wv4.w;
    }
    float score = 1.0f / (1.0f + __expf(-acc));
    float swb   = score + e_bias[lane];

    // sum of top-2 within each group of 4 consecutive lanes
    float a = swb;
    float b = __shfl_xor(a, 1, 32);
    float c = __shfl_xor(a, 2, 32);
    float d = __shfl_xor(a, 3, 32);
    float hi1 = fmaxf(a, b), lo1 = fminf(a, b);
    float hi2 = fmaxf(c, d), lo2 = fminf(c, d);
    float gs  = fmaxf(hi1, hi2) + fmaxf(fminf(hi1, hi2), fmaxf(lo1, lo2));

    // rank my group among the 8 group scores (earlier index wins ties)
    const int g = lane >> 2;
    int grank = 0;
#pragma unroll
    for (int j = 0; j < G; ++j) {
        float gj = __shfl(gs, j * 4, 32);
        grank += (gj > gs) || (gj == gs && j < g) ? 1 : 0;
    }
    float masked = (grank < TOPK_GROUP) ? swb : -3.0e38f;

    // top-8 of 32 masked scores
    int rank = 0;
#pragma unroll
    for (int j = 0; j < 32; ++j) {
        float mj = __shfl(masked, j, 32);
        rank += (mj > masked) || (mj == masked && j < lane) ? 1 : 0;
    }
    bool sel = (rank < TOP_K);
    float sc = sel ? score : 0.f;
    float denom = sc;
#pragma unroll
    for (int off = 16; off >= 1; off >>= 1) denom += __shfl_xor(denom, off, 32);
    float cw = sc / (denom + 1e-20f) * ROUTED_SCALE;

    if (sel) {
        int pos = atomicAdd(&cnt[lane], 1);
        tok_idx[lane * T + pos] = t;
        tok_w[lane * T + pos]   = cw;
    }
}

// ---------------- kernel 2: gated up-projection GEMM ----------------
// C[64 tokens x 64 cols of N] per block; 8 waves as 4(M) x 2(N);
// each wave: 16x32 for both gate and up mats -> 4 wmma per K-step.
// act = silu(x@Wg) * (x@Wu) stored bf16.
__global__ void up_gemm_kernel(const float* __restrict__ X,
                               const float* __restrict__ Wg,
                               const float* __restrict__ Wu,
                               unsigned short* __restrict__ Act,
                               const int* __restrict__ cnt,
                               const int* __restrict__ gidx,
                               int N, int K) {
    const int e  = blockIdx.z;
    const int m0 = blockIdx.y * 64;
    const int n0 = blockIdx.x * 64;
    const int cntE = cnt ? cnt[e] : T;
    if (m0 >= cntE) return;

    const float* WgE = Wg + (size_t)e * K * N;
    const float* WuE = Wu + (size_t)e * K * N;
    unsigned short* ActE = Act + (size_t)e * T * N;
    const int* gidxE = gidx ? gidx + e * T : nullptr;

    __shared__ __align__(16) unsigned int   lA[64 * 16];    // 64 rows x 32 bf16 (packed)
    __shared__ __align__(16) unsigned short lBg[64 * 32];   // [n][k] transposed bf16
    __shared__ __align__(16) unsigned short lBu[64 * 32];

    const int lane   = threadIdx.x & 31;
    const int wave   = threadIdx.x >> 5;
    const int wm     = wave >> 1;       // 0..3
    const int wn     = wave & 1;        // 0..1
    const int lanehi = lane >> 4;       // 0/1
    const int lanelo = lane & 15;

    v8f acc_g[2] = {};
    v8f acc_u[2] = {};

    // staging thread mapping
    const int ar = threadIdx.x >> 2;            // A: row 0..63
    const int ac = (threadIdx.x & 3) * 8;       // A: col 0,8,16,24
    const int bk = threadIdx.x >> 3;            // B: k-row 0..31
    const int bn = (threadIdx.x & 7) * 8;       // B: n 0..56

    int arow;
    {
        int m = m0 + ar;
        if (gidxE) arow = (m < cntE) ? gidxE[m] : 0;
        else       arow = m;
    }

    for (int k0 = 0; k0 < K; k0 += 32) {
        // ---- stage A (x rows, fp32 -> packed bf16 via packed cvt) ----
        {
            const float* src = X + (size_t)arow * K + k0 + ac;
            float4 f0 = *(const float4*)(src);
            float4 f1 = *(const float4*)(src + 4);
            uint2 p0 = pack4(f0);
            uint2 p1 = pack4(f1);
            uint4 p = make_uint4(p0.x, p0.y, p1.x, p1.y);
            *(uint4*)&lA[ar * 16 + (threadIdx.x & 3) * 4] = p;
        }
        // ---- stage Bg / Bu (fp32 row-major -> bf16 transposed [n][k]) ----
        {
            const float* srcg = WgE + (size_t)(k0 + bk) * N + n0 + bn;
            const float* srcu = WuE + (size_t)(k0 + bk) * N + n0 + bn;
            if (k0 + 32 < K) {
                __builtin_prefetch(srcg + (size_t)32 * N, 0, 1);
                __builtin_prefetch(srcu + (size_t)32 * N, 0, 1);
            }
            float4 g0 = *(const float4*)(srcg);
            float4 g1 = *(const float4*)(srcg + 4);
            float4 u0 = *(const float4*)(srcu);
            float4 u1 = *(const float4*)(srcu + 4);
            lBg[(bn + 0) * 32 + bk] = f2bf(g0.x);
            lBg[(bn + 1) * 32 + bk] = f2bf(g0.y);
            lBg[(bn + 2) * 32 + bk] = f2bf(g0.z);
            lBg[(bn + 3) * 32 + bk] = f2bf(g0.w);
            lBg[(bn + 4) * 32 + bk] = f2bf(g1.x);
            lBg[(bn + 5) * 32 + bk] = f2bf(g1.y);
            lBg[(bn + 6) * 32 + bk] = f2bf(g1.z);
            lBg[(bn + 7) * 32 + bk] = f2bf(g1.w);
            lBu[(bn + 0) * 32 + bk] = f2bf(u0.x);
            lBu[(bn + 1) * 32 + bk] = f2bf(u0.y);
            lBu[(bn + 2) * 32 + bk] = f2bf(u0.z);
            lBu[(bn + 3) * 32 + bk] = f2bf(u0.w);
            lBu[(bn + 4) * 32 + bk] = f2bf(u1.x);
            lBu[(bn + 5) * 32 + bk] = f2bf(u1.y);
            lBu[(bn + 6) * 32 + bk] = f2bf(u1.z);
            lBu[(bn + 7) * 32 + bk] = f2bf(u1.w);
        }
        __syncthreads();

        v16bf afrag = load_frag_a_u32(lA, wm * 16 + lanelo, lanehi);
        const int kb = lanehi * 16;
#pragma unroll
        for (int tt = 0; tt < 2; ++tt) {
            int n_l = wn * 32 + tt * 16 + lanelo;
            v16bf bg = load_frag_b(lBg, n_l, kb);
            acc_g[tt] = __builtin_amdgcn_wmma_f32_16x16x32_bf16(
                false, afrag, false, bg, (short)0, acc_g[tt], false, false);
            v16bf bu = load_frag_b(lBu, n_l, kb);
            acc_u[tt] = __builtin_amdgcn_wmma_f32_16x16x32_bf16(
                false, afrag, false, bu, (short)0, acc_u[tt], false, false);
        }
        __syncthreads();
    }

    // epilogue: act = silu(h1) * h2, store bf16
#pragma unroll
    for (int tt = 0; tt < 2; ++tt) {
        AccView hg, hu;
        hg.v = acc_g[tt];
        hu.v = acc_u[tt];
        const int n = n0 + wn * 32 + tt * 16 + lanelo;
#pragma unroll
        for (int j = 0; j < 8; ++j) {
            int m = m0 + wm * 16 + j + lanehi * 8;
            float act = siluf(hg.f[j]) * hu.f[j];
            ActE[(size_t)m * N + n] = f2bf(act);
        }
    }
}

// ---------------- kernel 3: down-projection GEMM ----------------
// A = bf16 act [slots x K]; B = fp32 w_down [K x N]; out fp32 [T x H].
// A tile is raw bf16 in global memory -> staged by the Tensor Data Mover
// (TDM descriptor per ISA ch.8: 2D tile 64x32, data_size=2B, dim0 stride=K),
// falling back to per-lane async-to-LDS, then to plain loads.
__global__ void down_gemm_kernel(const unsigned short* __restrict__ Act,
                                 const float* __restrict__ Wd,
                                 float* __restrict__ out,
                                 const int* __restrict__ cnt,
                                 const int* __restrict__ tok_idx,
                                 const float* __restrict__ tok_w,
                                 int N, int K) {
    const int e  = blockIdx.z;
    const int m0 = blockIdx.y * 64;
    const int n0 = blockIdx.x * 64;
    const int cntE = cnt ? cnt[e] : T;
    if (m0 >= cntE) return;

    const unsigned short* ActE = Act + (size_t)e * T * K;
    const float* WdE = Wd + (size_t)e * K * N;
    const int* tokE  = tok_idx ? tok_idx + e * T : nullptr;
    const float* twE = tok_w ? tok_w + e * T : nullptr;

    __shared__ __align__(16) unsigned short lA[64 * 32];   // act tile bf16 [r][k]
    __shared__ __align__(16) unsigned short lB[64 * 32];   // [n][k] transposed bf16

    const int lane   = threadIdx.x & 31;
    const int wave   = threadIdx.x >> 5;
    const int wm     = wave >> 1;
    const int wn     = wave & 1;
    const int lanehi = lane >> 4;
    const int lanelo = lane & 15;

    v8f acc[2] = {};

    const int ar = threadIdx.x >> 2;        // A row 0..63
    const int as = (threadIdx.x & 3) * 8;   // A k-seg (8 bf16 = 16B)
    const int bk = threadIdx.x >> 3;
    const int bn = (threadIdx.x & 7) * 8;

    for (int k0 = 0; k0 < K; k0 += 32) {
        // ---- stage A: raw bf16 copy global -> LDS ----
#if HAVE_TDM
        if (wave == 0) {
            const unsigned short* src = ActE + (size_t)m0 * K + k0;
            unsigned long long ga = (unsigned long long)(uintptr_t)src;
            unsigned lds_off = (unsigned)(uintptr_t)(&lA[0]);
            // D# group0: count=1, lds_addr, global_addr, type=2 (bits 127:126)
            u32x4 g0 = { 1u, lds_off, (unsigned)ga,
                         (unsigned)(((ga >> 32) & 0x01FFFFFFull) | 0x80000000u) };
            // D# group1: data_size=2B (bits17:16=1); tensor_dim0=32 (bits79:48);
            // tensor_dim1=64 (bits111:80); tile_dim0=32 (127:112); tile_dim1=64 (143:128);
            // tensor_dim0_stride=K (207:160)
            i32x8 g1 = { 0x00010000, (int)(32u << 16), (int)(64u << 16),
                         (int)(32u << 16), 64, K, 0, 0 };
            i32x4 z4 = { 0, 0, 0, 0 };
#if __clang_major__ >= 23
            i32x8 z8 = { 0, 0, 0, 0, 0, 0, 0, 0 };
            __builtin_amdgcn_tensor_load_to_lds(g0, g1, z4, z4, z8, 0);
#else
            __builtin_amdgcn_tensor_load_to_lds(g0, g1, z4, z4, 0);
#endif
        }
#else
        {
            const unsigned short* src = ActE + (size_t)(m0 + ar) * K + k0 + as;
#if USE_ASYNC_LDS
            unsigned lds_off = (unsigned)(unsigned long long)(uintptr_t)(&lA[ar * 32 + as]);
            asm volatile("global_load_async_to_lds_b128 %0, %1, off"
                         :: "v"(lds_off), "v"(src) : "memory");
#else
            *(uint4*)&lA[ar * 32 + as] = *(const uint4*)src;
#endif
        }
#endif
        // ---- stage B (fp32 -> bf16 transposed) ----
        {
            const float* srcb = WdE + (size_t)(k0 + bk) * N + n0 + bn;
            if (k0 + 32 < K) __builtin_prefetch(srcb + (size_t)32 * N, 0, 1);
            float4 b0 = *(const float4*)(srcb);
            float4 b1 = *(const float4*)(srcb + 4);
            lB[(bn + 0) * 32 + bk] = f2bf(b0.x);
            lB[(bn + 1) * 32 + bk] = f2bf(b0.y);
            lB[(bn + 2) * 32 + bk] = f2bf(b0.z);
            lB[(bn + 3) * 32 + bk] = f2bf(b0.w);
            lB[(bn + 4) * 32 + bk] = f2bf(b1.x);
            lB[(bn + 5) * 32 + bk] = f2bf(b1.y);
            lB[(bn + 6) * 32 + bk] = f2bf(b1.z);
            lB[(bn + 7) * 32 + bk] = f2bf(b1.w);
        }
#if HAVE_TDM
        if (wave == 0) __builtin_amdgcn_s_wait_tensorcnt(0);
#elif USE_ASYNC_LDS
        asm volatile("s_wait_asynccnt 0" ::: "memory");
#endif
        __syncthreads();

        v16bf afrag = load_frag_a_u16(lA, wm * 16 + lanelo, lanehi);
        const int kb = lanehi * 16;
#pragma unroll
        for (int tt = 0; tt < 2; ++tt) {
            v16bf bf = load_frag_b(lB, wn * 32 + tt * 16 + lanelo, kb);
            acc[tt] = __builtin_amdgcn_wmma_f32_16x16x32_bf16(
                false, afrag, false, bf, (short)0, acc[tt], false, false);
        }
        __syncthreads();
    }

    // epilogue
#pragma unroll
    for (int tt = 0; tt < 2; ++tt) {
        AccView av;
        av.v = acc[tt];
        const int n = n0 + wn * 32 + tt * 16 + lanelo;
#pragma unroll
        for (int j = 0; j < 8; ++j) {
            int m = m0 + wm * 16 + j + lanehi * 8;
            if (tokE) {
                if (m < cntE) {
                    int tok = tokE[m];
                    unsafeAtomicAdd(&out[(size_t)tok * H + n], av.f[j] * twE[m]);
                }
            } else {
                out[(size_t)m * H + n] = av.f[j];
            }
        }
    }
}

// ---------------- host launch ----------------
extern "C" void kernel_launch(void* const* d_in, const int* in_sizes, int n_in,
                              void* d_out, int out_size, void* d_ws, size_t ws_size,
                              hipStream_t stream) {
    (void)in_sizes; (void)n_in; (void)out_size; (void)ws_size;
    const float* x       = (const float*)d_in[0];   // [T,H]
    const float* gate_w  = (const float*)d_in[1];   // [E,H]
    const float* e_bias  = (const float*)d_in[2];   // [E]
    const float* w_gate  = (const float*)d_in[3];   // [E,H,I]
    const float* w_up    = (const float*)d_in[4];   // [E,H,I]
    const float* w_down  = (const float*)d_in[5];   // [E,I,H]
    const float* ws_gate = (const float*)d_in[6];   // [H,IS]
    const float* ws_up   = (const float*)d_in[7];   // [H,IS]
    const float* ws_down = (const float*)d_in[8];   // [IS,H]
    float* out = (float*)d_out;                     // [T,H]

    char* ws = (char*)d_ws;
    int*   cnt     = (int*)(ws + OFF_CNT);
    int*   tok_idx = (int*)(ws + OFF_TIDX);
    float* tok_w   = (float*)(ws + OFF_TW);
    unsigned short* act_r = (unsigned short*)(ws + OFF_ACTR);  // [E,T,I]
    unsigned short* act_s = (unsigned short*)(ws + OFF_ACTS);  // [T,IS]

    moe_init_kernel<<<1, 32, 0, stream>>>(cnt);
    gate_route_kernel<<<T / 8, 256, 0, stream>>>(x, gate_w, e_bias, cnt, tok_idx, tok_w);

    // shared expert: writes out directly (down stores), then routed adds on top
    up_gemm_kernel<<<dim3(IS / 64, T / 64, 1), 256, 0, stream>>>(
        x, ws_gate, ws_up, act_s, nullptr, nullptr, IS, H);
    down_gemm_kernel<<<dim3(H / 64, T / 64, 1), 256, 0, stream>>>(
        act_s, ws_down, out, nullptr, nullptr, nullptr, H, IS);

    // routed experts (sparse, gathered token lists)
    up_gemm_kernel<<<dim3(I / 64, T / 64, E), 256, 0, stream>>>(
        x, w_gate, w_up, act_r, cnt, tok_idx, I, H);
    down_gemm_kernel<<<dim3(H / 64, T / 64, E), 256, 0, stream>>>(
        act_r, w_down, out, cnt, tok_idx, tok_w, H, I);
}
